// VariableLengthAttention_80041010528397
// MI455X (gfx1250) — matrix operand order repaired and checked by reading, not compiled, not run on hardware
//
#include <hip/hip_runtime.h>
#include <hip/hip_bf16.h>

typedef __attribute__((ext_vector_type(16))) _Float16 v16h;
typedef __attribute__((ext_vector_type(8)))  _Float16 v8h;
typedef __attribute__((ext_vector_type(8)))  float    v8f;
typedef __attribute__((ext_vector_type(4)))  float    v4f;

#define NB     16
#define T_MEL  2048
#define T_TXT  1024
#define DM     512

// ---- LDS layout for the fused attention kernel (dynamic shared) ----
#define SSTRIDE (T_MEL + 4)     // f32 score row stride (pad to dodge bank conflicts)
#define QSTRIDE (DM + 8)        // f16 Q row stride
#define PSTRIDE (T_MEL + 8)     // f16 prob row stride
#define OSTRIDE (DM + 4)        // f32 O row stride (aliases S buffer)

static constexpr size_t SMEM_S = (size_t)16 * SSTRIDE * sizeof(float);     // 131328
static constexpr size_t SMEM_Q = (size_t)16 * QSTRIDE * sizeof(_Float16);  // 16640
static constexpr size_t SMEM_P = (size_t)16 * PSTRIDE * sizeof(_Float16);  // 65792
static constexpr size_t SMEM_M = (size_t)T_MEL;                            // 2048
static constexpr size_t SMEM_TOTAL = SMEM_S + SMEM_Q + SMEM_P + SMEM_M;    // 215808 < 320K

// A-fragment (16x32 f16) from an f16 row-major tile: lane l -> row l%16,
// contiguous K = kb + (l/16)*16 .. +15 (matches contiguous B fragments).
__device__ __forceinline__ v16h a_frag_f16(const _Float16* base, int stride, int kb, int lane) {
    const _Float16* p = base + (size_t)(lane & 15) * stride + kb + ((lane >> 4) << 4);
    v8h lo = *(const v8h*)p;
    v8h hi = *(const v8h*)(p + 8);
    v16h a;
#pragma unroll
    for (int i = 0; i < 8; ++i) { a[i] = lo[i]; a[8 + i] = hi[i]; }
    return a;
}

// ---------------- Weight transpose + f16 convert: WT[o][i] = W[i][o] ----------------
__global__ __launch_bounds__(256) void wt_prep_kernel(
    const float* __restrict__ Wq, const float* __restrict__ Wk, const float* __restrict__ Wv,
    _Float16* __restrict__ WqT, _Float16* __restrict__ WkT, _Float16* __restrict__ WvT)
{
    const float* W  = (blockIdx.y == 0) ? Wq  : (blockIdx.y == 1) ? Wk  : Wv;
    _Float16*   WT  = (blockIdx.y == 0) ? WqT : (blockIdx.y == 1) ? WkT : WvT;
    int o = blockIdx.x;
    for (int i = threadIdx.x; i < DM; i += 256)
        WT[(size_t)o * DM + i] = (_Float16)W[(size_t)i * DM + o];
}

// ---------------- Projection GEMM: Out_f16 = X_f32[M,512] @ W ----------------
// Each wave computes a 16x64 tile (A fragment reused across 4 WMMAs), with the
// next k-step's A/B fragments prefetched while the current WMMAs run.
// out_transposed==1 stores V^T layout: VT[(b*512 + d) * 2048 + s], row = b*2048+s.
__global__ __launch_bounds__(256) void proj_kernel(
    const float* __restrict__ X, const _Float16* __restrict__ WT,
    _Float16* __restrict__ Out, int M, int out_transposed)
{
    const int lane = threadIdx.x & 31;
    const int wid  = (blockIdx.x * 256 + threadIdx.x) >> 5;
    const int m0 = (wid >> 3) * 16;
    const int n0 = (wid & 7) * 64;
    if (m0 >= M) return;

    const int arow = m0 + (lane & 15);
    const int aoff = (lane >> 4) << 4;
    const int koff = (lane >> 4) << 4;
    const float* Xrow = X + (size_t)arow * DM + aoff;

    auto load_a = [&](int kb) -> v16h {
        const float* p = Xrow + kb;
        v4f x0 = *(const v4f*)(p);
        v4f x1 = *(const v4f*)(p + 4);
        v4f y0 = *(const v4f*)(p + 8);
        v4f y1 = *(const v4f*)(p + 12);
        v16h a;
#pragma unroll
        for (int i = 0; i < 4; ++i) {
            a[i]      = (_Float16)x0[i];
            a[4 + i]  = (_Float16)x1[i];
            a[8 + i]  = (_Float16)y0[i];
            a[12 + i] = (_Float16)y1[i];
        }
        return a;
    };

    v8f acc[4] = {};
    v16h a0 = load_a(0);
    v16h bf0[4];
#pragma unroll
    for (int j = 0; j < 4; ++j)
        bf0[j] = *(const v16h*)(WT + (size_t)(n0 + j * 16 + (lane & 15)) * DM + koff);

    for (int kb = 0; kb < DM - 32; kb += 32) {
        v16h a1 = load_a(kb + 32);
        v16h bf1[4];
#pragma unroll
        for (int j = 0; j < 4; ++j)
            bf1[j] = *(const v16h*)(WT + (size_t)(n0 + j * 16 + (lane & 15)) * DM + kb + 32 + koff);
#pragma unroll
        for (int j = 0; j < 4; ++j)
            acc[j] = __builtin_amdgcn_wmma_f32_16x16x32_f16(
                false, a0, false, bf0[j], (short)0, acc[j], false, false);
        a0 = a1;
#pragma unroll
        for (int j = 0; j < 4; ++j) bf0[j] = bf1[j];
    }
#pragma unroll
    for (int j = 0; j < 4; ++j)
        acc[j] = __builtin_amdgcn_wmma_f32_16x16x32_f16(
            false, a0, false, bf0[j], (short)0, acc[j], false, false);

    const int rbase = m0 + ((lane >> 4) << 3);
#pragma unroll
    for (int j = 0; j < 4; ++j) {
        int col = n0 + j * 16 + (lane & 15);
#pragma unroll
        for (int r = 0; r < 8; ++r) {
            int row = rbase + r;
            _Float16 h = (_Float16)acc[j][r];
            if (!out_transposed) {
                Out[(size_t)row * DM + col] = h;
            } else {
                int bb = row >> 11;          // row / 2048
                int s  = row & (T_MEL - 1);
                Out[((size_t)bb * DM + col) * T_MEL + s] = h;
            }
        }
    }
}

// ---------------- Fused attention: scores -> softmax -> attn out -> P@V -> LayerNorm ----
__global__ __launch_bounds__(256) void attn_fused_kernel(
    const _Float16* __restrict__ Qh, const _Float16* __restrict__ Kh, const _Float16* __restrict__ VT,
    const unsigned char* __restrict__ mel_mask, const unsigned char* __restrict__ src_mask,
    const float* __restrict__ gamma, const float* __restrict__ beta,
    float* __restrict__ out, float* __restrict__ attn)
{
    extern __shared__ char smem[];
    float*         S    = (float*)smem;                                  // [16][SSTRIDE]
    _Float16*      Qs   = (_Float16*)(smem + SMEM_S);                    // [16][QSTRIDE]
    _Float16*      P    = (_Float16*)(smem + SMEM_S + SMEM_Q);          // [16][PSTRIDE]
    unsigned char* melm = (unsigned char*)(smem + SMEM_S + SMEM_Q + SMEM_P);
    float*         Of   = S;                                             // reuse, [16][OSTRIDE]

    const int tid  = threadIdx.x;
    const int lane = tid & 31;
    const int w    = tid >> 5;                 // 8 waves
    const int b    = blockIdx.x >> 6;          // 64 t-tiles per batch
    const int t0   = (blockIdx.x & 63) * 16;

    // stage Q tile [16][512] f16 and mel mask into LDS
    for (int idx = tid; idx < 16 * (DM / 8); idx += 256) {
        int r = idx >> 6, ch = idx & 63;
        *(uint4*)(Qs + (size_t)r * QSTRIDE + ch * 8) =
            *(const uint4*)(Qh + ((size_t)(b * T_TXT + t0 + r)) * DM + ch * 8);
    }
    for (int s = tid; s < T_MEL; s += 256) melm[s] = mel_mask[(size_t)b * T_MEL + s];
    __syncthreads();

    const float scale = 0.044194173824159216f; // 1/sqrt(512)

    // ---- scores S[16][2048] = (Q K^T) * scale ; pipelined K-fragment loads ----
    for (int st = w; st < T_MEL / 16; st += 8) {
        const int s0 = st * 16;
        const _Float16* kbase =
            Kh + ((size_t)(b * T_MEL + s0 + (lane & 15))) * DM + ((lane >> 4) << 4);
        if (st + 8 < T_MEL / 16)
            __builtin_prefetch(kbase + (size_t)8 * 16 * DM, 0, 1);  // next s-tile K rows

        v8f acc = {};
        v16h a0 = a_frag_f16(Qs, QSTRIDE, 0, lane);
        v16h b0 = *(const v16h*)(kbase);
        for (int kb = 0; kb < DM - 32; kb += 32) {
            v16h a1 = a_frag_f16(Qs, QSTRIDE, kb + 32, lane);
            v16h b1 = *(const v16h*)(kbase + kb + 32);
            acc = __builtin_amdgcn_wmma_f32_16x16x32_f16(
                false, a0, false, b0, (short)0, acc, false, false);
            a0 = a1; b0 = b1;
        }
        acc = __builtin_amdgcn_wmma_f32_16x16x32_f16(
            false, a0, false, b0, (short)0, acc, false, false);

        const int rb = (lane >> 4) << 3;
        const int sc = s0 + (lane & 15);
#pragma unroll
        for (int r = 0; r < 8; ++r)
            S[(size_t)(rb + r) * SSTRIDE + sc] = acc[r] * scale;
    }
    __syncthreads();

    // ---- masked softmax over s; attn -> global (non-temporal), probs -> LDS f16 ----
    {
        const int row = tid >> 4, c = tid & 15;
        const int rmask = src_mask[(size_t)b * T_TXT + t0 + row];
        float* Srow = S + (size_t)row * SSTRIDE;
        float m = -3.0e38f;
        for (int s = c; s < T_MEL; s += 16) {
            float v = Srow[s];
            if (rmask && melm[s]) v = -3.0e38f;
            m = fmaxf(m, v);
        }
#pragma unroll
        for (int o = 8; o >= 1; o >>= 1) m = fmaxf(m, __shfl_xor(m, o, 16));
        float sum = 0.f;
        for (int s = c; s < T_MEL; s += 16) {
            float v = Srow[s];
            float e = (rmask && melm[s]) ? 0.f : __expf(v - m);
            Srow[s] = e;
            sum += e;
        }
#pragma unroll
        for (int o = 8; o >= 1; o >>= 1) sum += __shfl_xor(sum, o, 16);
        const float inv = 1.f / sum;
        float* arow = attn + ((size_t)(b * T_TXT + t0 + row)) * T_MEL;
        _Float16* Prow = P + (size_t)row * PSTRIDE;
        for (int s = c; s < T_MEL; s += 16) {
            float p = Srow[s] * inv;
            __builtin_nontemporal_store(p, &arow[s]);   // 134MB stream: keep out of L2
            Prow[s] = (_Float16)p;
        }
    }
    __syncthreads();

    // ---- O[16][512] = P[16][2048] @ V ; B fragments from V^T (K-contiguous),
    //      4 B-fragments batched per step + next-step prefetch into registers ----
    {
        const int koff = (lane >> 4) << 4;
        const _Float16* vbase[4];
#pragma unroll
        for (int j = 0; j < 4; ++j)
            vbase[j] = VT + ((size_t)(b * DM + (w * 4 + j) * 16 + (lane & 15))) * T_MEL + koff;

        v8f acc[4] = {};
        v16h a0 = a_frag_f16(P, PSTRIDE, 0, lane);
        v16h bf0[4];
#pragma unroll
        for (int j = 0; j < 4; ++j) bf0[j] = *(const v16h*)(vbase[j]);

        for (int sb = 0; sb < T_MEL - 32; sb += 32) {
            v16h a1 = a_frag_f16(P, PSTRIDE, sb + 32, lane);
            v16h bf1[4];
#pragma unroll
            for (int j = 0; j < 4; ++j) bf1[j] = *(const v16h*)(vbase[j] + sb + 32);
#pragma unroll
            for (int j = 0; j < 4; ++j)
                acc[j] = __builtin_amdgcn_wmma_f32_16x16x32_f16(
                    false, a0, false, bf0[j], (short)0, acc[j], false, false);
            a0 = a1;
#pragma unroll
            for (int j = 0; j < 4; ++j) bf0[j] = bf1[j];
        }
#pragma unroll
        for (int j = 0; j < 4; ++j)
            acc[j] = __builtin_amdgcn_wmma_f32_16x16x32_f16(
                false, a0, false, bf0[j], (short)0, acc[j], false, false);

        const int rb = (lane >> 4) << 3;
#pragma unroll
        for (int j = 0; j < 4; ++j) {
            int col = (w * 4 + j) * 16 + (lane & 15);
#pragma unroll
            for (int r = 0; r < 8; ++r)
                Of[(size_t)(rb + r) * OSTRIDE + col] = acc[j][r];
        }
    }
    __syncthreads();

    // ---- LayerNorm over last dim (512), biased variance, eps=1e-5 ----
    {
        const int row = tid >> 4, c = tid & 15;
        const float* Orow = Of + (size_t)row * OSTRIDE;
        float s1 = 0.f, s2 = 0.f;
        for (int d = c; d < DM; d += 16) { float x = Orow[d]; s1 += x; s2 += x * x; }
#pragma unroll
        for (int o = 8; o >= 1; o >>= 1) { s1 += __shfl_xor(s1, o, 16); s2 += __shfl_xor(s2, o, 16); }
        const float mean = s1 * (1.f / DM);
        const float var  = s2 * (1.f / DM) - mean * mean;
        const float rstd = rsqrtf(var + 1e-5f);
        float* orow = out + ((size_t)(b * T_TXT + t0 + row)) * DM;
        for (int d = c; d < DM; d += 16) {
            float y = (Orow[d] - mean) * rstd * gamma[d] + beta[d];
            __builtin_nontemporal_store(y, &orow[d]);   // written once, never re-read
        }
    }
}

extern "C" void kernel_launch(void* const* d_in, const int* in_sizes, int n_in,
                              void* d_out, int out_size, void* d_ws, size_t ws_size,
                              hipStream_t stream) {
    const float*         mel      = (const float*)d_in[0];          // [16,2048,512]
    const float*         text     = (const float*)d_in[1];          // [16,1024,512]
    const unsigned char* mel_mask = (const unsigned char*)d_in[2];  // [16,2048] bool
    const unsigned char* src_mask = (const unsigned char*)d_in[3];  // [16,1024] bool
    const float*         Wq       = (const float*)d_in[4];          // [512,512]
    const float*         Wk       = (const float*)d_in[5];
    const float*         Wv       = (const float*)d_in[6];
    const float*         gamma    = (const float*)d_in[7];
    const float*         beta     = (const float*)d_in[8];

    float* out  = (float*)d_out;                                    // [16,1024,512]
    float* attn = out + (size_t)NB * T_TXT * DM;                    // [16,1024,2048]

    // workspace partition
    char* ws = (char*)d_ws;
    _Float16* WqT = (_Float16*)ws; ws += (size_t)DM * DM * 2;
    _Float16* WkT = (_Float16*)ws; ws += (size_t)DM * DM * 2;
    _Float16* WvT = (_Float16*)ws; ws += (size_t)DM * DM * 2;
    _Float16* Qh  = (_Float16*)ws; ws += (size_t)NB * T_TXT * DM * 2;
    _Float16* Kh  = (_Float16*)ws; ws += (size_t)NB * T_MEL * DM * 2;
    _Float16* VT  = (_Float16*)ws; ws += (size_t)NB * T_MEL * DM * 2;
    (void)ws_size; (void)in_sizes; (void)n_in; (void)out_size;

    wt_prep_kernel<<<dim3(DM, 3), 256, 0, stream>>>(Wq, Wk, Wv, WqT, WkT, WvT);

    // Q = text @ Wq  (M = 16*1024) ; K = mel @ Wk ; V^T = (mel @ Wv)^T  (M = 16*2048)
    proj_kernel<<<(NB * T_TXT) / 16, 256, 0, stream>>>(text, WqT, Qh, NB * T_TXT, 0);
    proj_kernel<<<(NB * T_MEL) / 16, 256, 0, stream>>>(mel,  WkT, Kh, NB * T_MEL, 0);
    proj_kernel<<<(NB * T_MEL) / 16, 256, 0, stream>>>(mel,  WvT, VT, NB * T_MEL, 1);

    hipFuncSetAttribute(reinterpret_cast<const void*>(attn_fused_kernel),
                        hipFuncAttributeMaxDynamicSharedMemorySize, (int)SMEM_TOTAL);
    attn_fused_kernel<<<NB * (T_TXT / 16), 256, SMEM_TOTAL, stream>>>(
        Qh, Kh, VT, mel_mask, src_mask, gamma, beta, out, attn);
}
